// PECircuits_29145648071275
// MI455X (gfx1250) — matrix operand order repaired and checked by reading, not compiled, not run on hardware
//
#include <hip/hip_runtime.h>

#define DIMN 2048
#define CTXN 512
#define BATCH 8192
#define NPV 512
#define NSV 256
#define NCAT (NPV + NSV)   // 768

typedef _Float16 f16t;
typedef __attribute__((ext_vector_type(16))) _Float16 v16h;
typedef __attribute__((ext_vector_type(8)))  _Float16 v8h;
typedef __attribute__((ext_vector_type(8)))  float    v8f;

// Pointer types matching __builtin_amdgcn_global_load_async_to_lds_b128:
// (v4i addrspace(1)* src, v4i addrspace(3)* dst, imm offset, imm cpol)
typedef int pe_v4i __attribute__((vector_size(16)));
typedef __attribute__((address_space(1))) pe_v4i* pe_gptr;
typedef __attribute__((address_space(3))) pe_v4i* pe_lptr;

enum { EP_PRED = 0, EP_RELU16 = 1, EP_RELU32 = 2, EP_SOM = 3, EP_POS = 4, EP_NEG = 5 };

#if defined(__gfx1250__) && __has_builtin(__builtin_amdgcn_global_load_async_to_lds_b128)
#define PE_ASYNC_LDS 1
#endif

#define LDS_PITCH 72   // 64 + 8 halves: 36-dword row pitch -> conflict-free b128 reads

// ---------------------------------------------------------------------------
// fp32 -> fp16 conversion, optionally scattering into a strided (concatenated)
// destination: dst[r*ldd + coff + c] = (f16) src[r*cols + c]
// ---------------------------------------------------------------------------
__global__ void pe_cvt_f32_f16(const float* __restrict__ src, f16t* __restrict__ dst,
                               long rows, long cols, long ldd, long coff) {
  long n = rows * cols;
  long stride = (long)gridDim.x * blockDim.x;
  if (ldd == cols && coff == 0) {
    for (long i = (long)blockIdx.x * blockDim.x + threadIdx.x; i < n; i += stride)
      dst[i] = (f16t)src[i];
  } else {
    for (long i = (long)blockIdx.x * blockDim.x + threadIdx.x; i < n; i += stride) {
      long r = i / cols;
      long c = i - r * cols;
      dst[r * ldd + coff + c] = (f16t)src[i];
    }
  }
}

// ---------------------------------------------------------------------------
// Generic WMMA GEMM:  out = epilogue( A(BxK) @ W(NxK)^T )
// f16 inputs, fp32 accumulate via v_wmma_f32_16x16x32_f16 (wave32).
// Block: 256 threads = 8 waves; wave w owns rows [blk.x*128 + w*16, +16),
// cols [blk.y*64, +64) as 4 v8f accumulators.
// W is double-buffer staged into LDS via global_load_async_to_lds_b128
// (ASYNCcnt-tracked), shared by all 8 waves.
// ---------------------------------------------------------------------------
template<int MODE>
__global__ __launch_bounds__(256, 2)
void pe_gemm_wmma(const f16t* __restrict__ A, int lda,
                  const f16t* __restrict__ W, int ldw,
                  int K, int N,
                  float* __restrict__ out32, f16t* __restrict__ out16, int ldo,
                  const float* __restrict__ bias,
                  const float* __restrict__ aux0,   // x (POS/NEG) | relu(s_in) (SOM)
                  const f16t*  __restrict__ aux1h,  // predicted f16 (POS/NEG)
                  const float* __restrict__ aux2)   // r_pos / r_neg (POS/NEG)
{
  __shared__ f16t sW[2][64 * LDS_PITCH];

  const int tid  = threadIdx.x;
  const int lane = tid & 31;
  const int wave = tid >> 5;
  const int hs   = lane >> 4;       // half-wave select
  const int lr   = lane & 15;
  const int m0   = (blockIdx.x * 8 + wave) * 16;
  const int n0   = blockIdx.y * 64;

  v8f acc[4] = {};

  const f16t* aPtr = A + (size_t)(m0 + lr) * lda + (hs << 3);

  // Cooperative staging of W[n0..n0+63, kc..kc+63]: 512 x 16B chunks,
  // 2 rounds of 256 lanes. round r, thread t -> idx = r*256+t,
  // row = idx/8, col halves = (idx%8)*8.
  auto stage = [&](int buf, int kc) {
#pragma unroll
    for (int r = 0; r < 2; ++r) {
      const int idx = (r << 8) + tid;
      const int row = idx >> 3;
      const int col = (idx & 7) << 3;
      const f16t* g = W + (size_t)(n0 + row) * ldw + (kc + col);
      f16t* l = &sW[buf][row * LDS_PITCH + col];
#ifdef PE_ASYNC_LDS
      __builtin_amdgcn_global_load_async_to_lds_b128((pe_gptr)g, (pe_lptr)l, 0, 0);
#else
      *(v8h*)l = *(const v8h*)g;
#endif
    }
  };
  auto stage_fence = [&]() {
#ifdef PE_ASYNC_LDS
    asm volatile("s_wait_asynccnt 0x0" ::: "memory");
#endif
  };

  // Consume one 64-wide K chunk from LDS buffer `buf`.
  // All 8 ds_load_b128 are issued into distinct registers BEFORE the WMMA
  // sequence so the dscnt waits can be staggered instead of serialized.
  auto compute = [&](int buf, int kc) {
#pragma unroll
    for (int kk = 0; kk < 64; kk += 32) {
      const v8h alo = *(const v8h*)(aPtr + kc + kk);        // K+{0..7} / {8..15}
      const v8h ahi = *(const v8h*)(aPtr + kc + kk + 16);   // K+{16..23} / {24..31}
      v16h a;
#pragma unroll
      for (int i = 0; i < 8; ++i) { a[i] = alo[i]; a[i + 8] = ahi[i]; }
      __builtin_prefetch(aPtr + kc + kk + 256, 0, 3);       // global_prefetch_b8

      v16h b[4];
#pragma unroll
      for (int t = 0; t < 4; ++t) {
        const f16t* bp = &sW[buf][(t * 16 + lr) * LDS_PITCH + (hs << 3) + kk];
        const v8h blo = *(const v8h*)bp;                    // ds_load_b128
        const v8h bhi = *(const v8h*)(bp + 16);             // ds_load_b128
#pragma unroll
        for (int i = 0; i < 8; ++i) { b[t][i] = blo[i]; b[t][i + 8] = bhi[i]; }
      }
#pragma unroll
      for (int t = 0; t < 4; ++t)
        acc[t] = __builtin_amdgcn_wmma_f32_16x16x32_f16(
            false, a, false, b[t], (short)0, acc[t], false, false);
    }
  };

  // Double-buffered main loop over K in 64-half chunks.
  int buf = 0;
  stage(0, 0);
  stage_fence();
  __syncthreads();
  for (int kc = 0; kc < K; kc += 64) {
    const bool more = (kc + 64) < K;
    if (more) stage(buf ^ 1, kc + 64);   // prefetch next chunk into other buffer
    compute(buf, kc);
    if (more) {
      stage_fence();
      __syncthreads();
      buf ^= 1;
    }
  }

  // Fused epilogues. C layout: elem v -> row = 8*(lane/16)+v, col = lane%16.
#pragma unroll
  for (int t = 0; t < 4; ++t) {
    const int col = n0 + t * 16 + lr;
#pragma unroll
    for (int v = 0; v < 8; ++v) {
      const int row = m0 + (hs << 3) + v;
      const float val = acc[t][v];
      const size_t oidx = (size_t)row * ldo + col;   // output (possibly strided)
      const size_t xidx = (size_t)row * N + col;     // aux tensors, dense geometry
      if (MODE == EP_PRED) {
        out16[oidx] = (f16t)(val + bias[col]);
      } else if (MODE == EP_RELU16) {
        out16[oidx] = (f16t)fmaxf(val, 0.0f);
      } else if (MODE == EP_RELU32) {
        out32[oidx] = fmaxf(val, 0.0f);
      } else if (MODE == EP_SOM) {
        out16[oidx] = (f16t)fmaxf(aux0[xidx] - val, 0.0f);  // relu(relu(s_in) - inh)
      } else if (MODE == EP_POS) {
        const float pe = fmaxf(aux0[xidx] - (float)aux1h[xidx] - val, 0.0f);
        out32[oidx] = 0.9f * aux2[xidx] + 0.1f * pe;
      } else { // EP_NEG
        const float pe = fmaxf((float)aux1h[xidx] - aux0[xidx] - val, 0.0f);
        out32[oidx] = 0.9f * aux2[xidx] + 0.1f * pe;
      }
    }
  }
}

// ---------------------------------------------------------------------------
extern "C" void kernel_launch(void* const* d_in, const int* in_sizes, int n_in,
                              void* d_out, int out_size, void* d_ws, size_t ws_size,
                              hipStream_t stream) {
  (void)in_sizes; (void)n_in; (void)out_size; (void)ws_size;
  const float* x     = (const float*)d_in[0];
  const float* ctx   = (const float*)d_in[1];
  const float* rpos  = (const float*)d_in[2];
  const float* rneg  = (const float*)d_in[3];
  const float* Wpred = (const float*)d_in[4];
  const float* bpred = (const float*)d_in[5];
  const float* Wpv1  = (const float*)d_in[6];
  const float* Wpv2  = (const float*)d_in[7];
  const float* Wvip  = (const float*)d_in[8];
  const float* Wsom  = (const float*)d_in[9];
  const float* Wvs   = (const float*)d_in[10];
  const float* Wpos  = (const float*)d_in[11];
  const float* Wneg  = (const float*)d_in[12];
  const float* Wden  = (const float*)d_in[13];

  char* ws = (char*)d_ws;
  size_t off = 0;
  auto carve = [&](size_t bytes) -> void* {
    void* p = ws + off;
    off += (bytes + 255) & ~(size_t)255;
    return p;
  };
  f16t*  xh     = (f16t*)carve((size_t)BATCH * DIMN * 2);
  f16t*  ch     = (f16t*)carve((size_t)BATCH * CTXN * 2);
  f16t*  predh  = (f16t*)carve((size_t)BATCH * DIMN * 2);
  f16t*  rpv2h  = (f16t*)carve((size_t)BATCH * NPV * 2);
  f16t*  rviph  = (f16t*)carve((size_t)BATCH * NSV * 2);
  float* sin32  = (float*)carve((size_t)BATCH * NSV * 4);
  f16t*  acat   = (f16t*)carve((size_t)BATCH * NCAT * 2);  // [r_pv1 | r_som]
  f16t*  wpredh = (f16t*)carve((size_t)DIMN * CTXN * 2);
  f16t*  wpv1h  = (f16t*)carve((size_t)NPV * DIMN * 2);
  f16t*  wpv2h  = (f16t*)carve((size_t)NPV * DIMN * 2);
  f16t*  wviph  = (f16t*)carve((size_t)NSV * DIMN * 2);
  f16t*  wsomh  = (f16t*)carve((size_t)NSV * DIMN * 2);
  f16t*  wvsh   = (f16t*)carve((size_t)NSV * NSV * 2);
  f16t*  wposh  = (f16t*)carve((size_t)DIMN * NPV * 2);
  f16t*  wcath  = (f16t*)carve((size_t)DIMN * NCAT * 2);   // [W_inh_pe_neg | W_som_dend]

  auto cvt = [&](const float* s, f16t* d, long rows, long cols, long ldd, long coff) {
    long n = rows * cols;
    int blocks = (int)((n + 255) / 256);
    if (blocks > 4096) blocks = 4096;
    pe_cvt_f32_f16<<<dim3(blocks), dim3(256), 0, stream>>>(s, d, rows, cols, ldd, coff);
  };
  cvt(x,     xh,     BATCH, DIMN, DIMN, 0);
  cvt(ctx,   ch,     BATCH, CTXN, CTXN, 0);
  cvt(Wpred, wpredh, DIMN,  CTXN, CTXN, 0);
  cvt(Wpv1,  wpv1h,  NPV,   DIMN, DIMN, 0);
  cvt(Wpv2,  wpv2h,  NPV,   DIMN, DIMN, 0);
  cvt(Wvip,  wviph,  NSV,   DIMN, DIMN, 0);
  cvt(Wsom,  wsomh,  NSV,   DIMN, DIMN, 0);
  cvt(Wvs,   wvsh,   NSV,   NSV,  NSV,  0);
  cvt(Wpos,  wposh,  DIMN,  NPV,  NPV,  0);
  cvt(Wneg,  wcath,  DIMN,  NPV,  NCAT, 0);     // cols 0..511 of concat weights
  cvt(Wden,  wcath,  DIMN,  NSV,  NCAT, NPV);   // cols 512..767

  const dim3 blk(256);
  auto grid = [](int Ncols) { return dim3(BATCH / 128, Ncols / 64); };

  // predicted (f16 only) = ctx @ Wpred^T + b
  pe_gemm_wmma<EP_PRED><<<grid(DIMN), blk, 0, stream>>>(
      ch, CTXN, wpredh, CTXN, CTXN, DIMN,
      nullptr, predh, DIMN, bpred, nullptr, nullptr, nullptr);
  // r_pv1 = relu(x @ Wpv1^T)  -> acat[:, 0:512]
  pe_gemm_wmma<EP_RELU16><<<grid(NPV), blk, 0, stream>>>(
      xh, DIMN, wpv1h, DIMN, DIMN, NPV,
      nullptr, acat, NCAT, nullptr, nullptr, nullptr, nullptr);
  // r_pv2 = relu(pred @ Wpv2^T)
  pe_gemm_wmma<EP_RELU16><<<grid(NPV), blk, 0, stream>>>(
      predh, DIMN, wpv2h, DIMN, DIMN, NPV,
      nullptr, rpv2h, NPV, nullptr, nullptr, nullptr, nullptr);
  // r_vip = relu(pred @ Wvip^T)
  pe_gemm_wmma<EP_RELU16><<<grid(NSV), blk, 0, stream>>>(
      predh, DIMN, wviph, DIMN, DIMN, NSV,
      nullptr, rviph, NSV, nullptr, nullptr, nullptr, nullptr);
  // relu(s_in) = relu(x @ Wsom^T)  (fp32 scratch)
  pe_gemm_wmma<EP_RELU32><<<grid(NSV), blk, 0, stream>>>(
      xh, DIMN, wsomh, DIMN, DIMN, NSV,
      sin32, nullptr, NSV, nullptr, nullptr, nullptr, nullptr);
  // r_som = relu(relu(s_in) - r_vip @ Wvs^T)  -> acat[:, 512:768]
  pe_gemm_wmma<EP_SOM><<<grid(NSV), blk, 0, stream>>>(
      rviph, NSV, wvsh, NSV, NSV, NSV,
      nullptr, acat + NPV, NCAT, nullptr, sin32, nullptr, nullptr);
  // new_r_pos = 0.9*r_pos + 0.1*relu(x - pred - r_pv2 @ Wpos^T)
  pe_gemm_wmma<EP_POS><<<grid(DIMN), blk, 0, stream>>>(
      rpv2h, NPV, wposh, NPV, NPV, DIMN,
      (float*)d_out, nullptr, DIMN, nullptr, x, predh, rpos);
  // new_r_neg = 0.9*r_neg + 0.1*relu(pred - x - [r_pv1|r_som] @ [Wneg|Wden]^T)
  pe_gemm_wmma<EP_NEG><<<grid(DIMN), blk, 0, stream>>>(
      acat, NCAT, wcath, NCAT, NCAT, DIMN,
      (float*)d_out + (size_t)BATCH * DIMN, nullptr, DIMN, nullptr, x, predh, rneg);
}
